// GroupedQueryAttention_73443940761654
// MI455X (gfx1250) — matrix operand order repaired and checked by reading, not compiled, not run on hardware
//
#include <hip/hip_runtime.h>

// ---------------------------------------------------------------- constants
#define BB 2
#define SS 2048
#define DD 2048
#define HH 32
#define KVH 8
#define HDIM 64
#define REPQ 4      // HH / KVH

typedef __attribute__((ext_vector_type(16))) __bf16        v16bf;
typedef __attribute__((ext_vector_type(8)))  float         v8f;
typedef __attribute__((ext_vector_type(8)))  unsigned int  v8u;
typedef __attribute__((ext_vector_type(4)))  unsigned int  u32x4;
typedef __attribute__((ext_vector_type(8)))  int           i32x8;
typedef __attribute__((ext_vector_type(4)))  int           i32x4;

#if defined(__has_builtin)
#if __has_builtin(__builtin_amdgcn_tensor_load_to_lds) && __has_builtin(__builtin_amdgcn_s_wait_tensorcnt)
#define HAVE_TDM 1
#endif
#endif

__device__ __forceinline__ unsigned short f2bf(float f) {
    unsigned u = __float_as_uint(f);
    u += 0x7fffu + ((u >> 16) & 1u);      // round-to-nearest-even
    return (unsigned short)(u >> 16);
}

__device__ __forceinline__ v8f wmma_bf16(v8u a, v8u b, v8f c) {
    return __builtin_amdgcn_wmma_f32_16x16x32_bf16(
        /*neg_a=*/false, __builtin_bit_cast(v16bf, a),
        /*neg_b=*/false, __builtin_bit_cast(v16bf, b),
        /*c_mod=*/(short)0, c, /*reuse_a=*/false, /*reuse_b=*/false);
}

// ---------------------------------------------------------------- cast kernels
__global__ void cast_f32_bf16(const float* __restrict__ src,
                              unsigned short* __restrict__ dst, int n) {
    int i = blockIdx.x * blockDim.x + threadIdx.x;
    if (i < n) dst[i] = f2bf(src[i]);
}

// pack [wq | wk | wv] -> bf16 [DD][3072]
__global__ void pack_wqkv(const float* __restrict__ wq, const float* __restrict__ wk,
                          const float* __restrict__ wv, unsigned short* __restrict__ dst) {
    int i = blockIdx.x * blockDim.x + threadIdx.x;
    const int NT = 3072;
    if (i >= DD * NT) return;
    int r = i / NT, c = i % NT;
    float v;
    if (c < 2048)       v = wq[r * 2048 + c];
    else if (c < 2560)  v = wk[r * 512 + (c - 2048)];
    else                v = wv[r * 512 + (c - 2560)];
    dst[i] = f2bf(v);
}

// ---------------------------------------------------------------- GEMM (bf16 WMMA, TDM + double buffer)
// C[M,N] f32 = A[M,K] bf16 (row-major) * B[K,N] bf16 (row-major)
// M,N multiples of 128; K multiple of 32.
#define BM 128
#define BN 128
#define BK 32
__global__ __launch_bounds__(256)
void gemm_bf16(const unsigned short* __restrict__ A, const unsigned short* __restrict__ Bm,
               float* __restrict__ C, int M, int N, int K) {
    __shared__ unsigned int As[2][BM][18];        // (k,k+1) bf16 pairs: 16 data dwords + 2 pad
    __shared__ unsigned int Bs[2][16][BN + 4];    // [k-pair][col] packed pairs, pad 4

    const int tid  = threadIdx.x;
    const int lane = tid & 31;
    const int w    = tid >> 5;
    const int wrow = w & 3;           // 4 row groups of 32
    const int wcol = w >> 2;          // 2 col groups of 64
    const int l15  = lane & 15;
    const int hi   = lane >> 4;
    const int bm   = blockIdx.x * BM;
    const int bn   = blockIdx.y * BN;
    const int nk   = K / BK;

    v8f acc[2][4] = {};

    // per-thread B staging coords: fixed column, k-pair rows p0+4j
    const int p0 = tid >> 6;
    const int cc = (tid & 63) * 2;
    const unsigned short* bptr = Bm + (size_t)(2 * p0) * N + bn + cc;
    const size_t bstep = (size_t)8 * N;          // j stride in elements
    unsigned ta[4], tb[4];

#ifdef HAVE_TDM
    // Tensor DMA descriptor group1 (loop-invariant):
    //   data_size=2B, pad_enable, pad_interval=3 (16 dwords), pad_amount=1 (2 dwords)
    //   tensor_dim0=32, tensor_dim1=128, tile_dim0=32, tile_dim1=128, dim0_stride=K
    i32x8 g1;
    g1[0] = 0x02D10000;
    g1[1] = (32  << 16);
    g1[2] = (128 << 16);
    g1[3] = (32  << 16);
    g1[4] = 128;
    g1[5] = K;
    g1[6] = 0;
    g1[7] = 0;
    i32x4 gz = {0, 0, 0, 0};
    const unsigned lds_a[2] = { (unsigned)(size_t)&As[0][0][0], (unsigned)(size_t)&As[1][0][0] };
    const unsigned long long abase = (unsigned long long)(size_t)(A + (size_t)bm * K);
    auto tdm_issue = [&](int nb, int kk) {
        unsigned long long ga = abase + (unsigned long long)kk * 2u;
        u32x4 g0;
        g0[0] = 1u;                                   // count=1 (valid descriptor)
        g0[1] = lds_a[nb];                            // LDS byte address
        g0[2] = (unsigned)ga;                         // global_addr[31:0]
        g0[3] = (unsigned)(ga >> 32) | 0x80000000u;   // global_addr[56:32] | type=2
#if __clang_major__ >= 23
        i32x8 gz8 = {0, 0, 0, 0, 0, 0, 0, 0};
        __builtin_amdgcn_tensor_load_to_lds(g0, g1, gz, gz, gz8, 0);
#else
        __builtin_amdgcn_tensor_load_to_lds(g0, g1, gz, gz, 0);
#endif
    };
#else
    // fallback staging coords for A: fixed pair-col p, rows ar0 + 16i
    const int ar0 = tid >> 4;
    const int ap  = tid & 15;
    const unsigned short* aptr = A + (size_t)(bm + ar0) * K + 2 * ap;
    unsigned au[8];
#endif

    // ---- prologue: stage tile 0 into buffer 0
#ifdef HAVE_TDM
    if (w == 0) tdm_issue(0, 0);
#else
    #pragma unroll
    for (int i = 0; i < 8; ++i) au[i] = *(const unsigned*)(aptr + (size_t)i * 16 * K);
    #pragma unroll
    for (int i = 0; i < 8; ++i) As[0][ar0 + 16 * i][ap] = au[i];
    aptr += BK;
#endif
    #pragma unroll
    for (int j = 0; j < 4; ++j) {
        ta[j] = *(const unsigned*)(bptr + j * bstep);
        tb[j] = *(const unsigned*)(bptr + j * bstep + N);
    }
    bptr += (size_t)BK * N;
    #pragma unroll
    for (int j = 0; j < 4; ++j) {
        Bs[0][p0 + 4 * j][cc]     = (ta[j] & 0xffffu) | (tb[j] << 16);
        Bs[0][p0 + 4 * j][cc + 1] = (ta[j] >> 16)     | (tb[j] & 0xffff0000u);
    }
#ifdef HAVE_TDM
    if (w == 0) __builtin_amdgcn_s_wait_tensorcnt(0);
#endif
    __syncthreads();

    // ---- pipelined main loop: one barrier per K-step
    for (int ks = 0; ks < nk; ++ks) {
        const int buf  = ks & 1;
        const bool more = (ks + 1) < nk;

        if (more) {
#ifdef HAVE_TDM
            if (w == 0) tdm_issue(buf ^ 1, (ks + 1) * BK);
#else
            #pragma unroll
            for (int i = 0; i < 8; ++i) au[i] = *(const unsigned*)(aptr + (size_t)i * 16 * K);
            aptr += BK;
#endif
            #pragma unroll
            for (int j = 0; j < 4; ++j) {
                ta[j] = *(const unsigned*)(bptr + j * bstep);
                tb[j] = *(const unsigned*)(bptr + j * bstep + N);
            }
            bptr += (size_t)BK * N;
        }

        // fragments + 8 WMMAs from current buffer
        v8u af[2], bfr[4];
        #pragma unroll
        for (int mt = 0; mt < 2; ++mt) {
            int r  = wrow * 32 + mt * 16 + l15;
            int pb = hi * 8;
            #pragma unroll
            for (int v = 0; v < 4; ++v) {
                af[mt][v]     = As[buf][r][pb + v];
                af[mt][4 + v] = As[buf][r][4 + pb + v];
            }
        }
        #pragma unroll
        for (int nt = 0; nt < 4; ++nt) {
            int c   = wcol * 64 + nt * 16 + l15;
            int pb0 = hi * 8;
            #pragma unroll
            for (int v = 0; v < 8; ++v) bfr[nt][v] = Bs[buf][pb0 + v][c];
        }
        #pragma unroll
        for (int mt = 0; mt < 2; ++mt)
            #pragma unroll
            for (int nt = 0; nt < 4; ++nt)
                acc[mt][nt] = wmma_bf16(af[mt], bfr[nt], acc[mt][nt]);

        if (more) {
            // commit next B tile into the other buffer
            #pragma unroll
            for (int j = 0; j < 4; ++j) {
                Bs[buf ^ 1][p0 + 4 * j][cc]     = (ta[j] & 0xffffu) | (tb[j] << 16);
                Bs[buf ^ 1][p0 + 4 * j][cc + 1] = (ta[j] >> 16)     | (tb[j] & 0xffff0000u);
            }
#ifdef HAVE_TDM
            if (w == 0) __builtin_amdgcn_s_wait_tensorcnt(0);
#else
            #pragma unroll
            for (int i = 0; i < 8; ++i) As[buf ^ 1][ar0 + 16 * i][ap] = au[i];
#endif
        }
        __syncthreads();
    }

    // ---- epilogue: pointer-walk stores (immediate column offsets)
    float* cp = C + (size_t)(bm + wrow * 32 + hi * 8) * N + bn + wcol * 64 + l15;
    #pragma unroll
    for (int mt = 0; mt < 2; ++mt) {
        float* cr = cp + (size_t)(mt * 16) * N;
        #pragma unroll
        for (int i = 0; i < 8; ++i) {
            #pragma unroll
            for (int nt = 0; nt < 4; ++nt) cr[nt * 16] = acc[mt][nt][i];
            cr += N;
        }
    }
}

// ---------------------------------------------------------------- RoPE (q,k) from fused qkv f32
// qkv: [B,S,3072] f32.  Writes Qb [B,H,S,HD] bf16 and Kb [B,KV,S,HD] bf16.
__global__ void rope_qk(const float* __restrict__ qkv,
                        unsigned short* __restrict__ Qb,
                        unsigned short* __restrict__ Kb) {
    int idx = blockIdx.x * blockDim.x + threadIdx.x;
    const int HT = HH + KVH;                       // 40
    if (idx >= BB * SS * HT * 32) return;
    int i  = idx & 31;
    int hh = (idx >> 5) % HT;
    int s  = (idx / (32 * HT)) % SS;
    int b  = idx / (32 * HT * SS);

    float inv = __expf(-(float)(2 * i) * (9.210340371976184f / 64.0f)); // theta^{-2i/64}
    float ang = (float)s * inv;
    float sn, cs;
    __sincosf(ang, &sn, &cs);

    const float* row = qkv + ((size_t)b * SS + s) * 3072;
    if (hh < HH) {
        float x1 = row[hh * 64 + i];
        float x2 = row[hh * 64 + i + 32];
        unsigned short* o = Qb + (((size_t)b * HH + hh) * SS + s) * HDIM;
        o[i]      = f2bf(x1 * cs - x2 * sn);
        o[i + 32] = f2bf(x1 * sn + x2 * cs);
    } else {
        int g = hh - HH;
        float x1 = row[2048 + g * 64 + i];
        float x2 = row[2048 + g * 64 + i + 32];
        unsigned short* o = Kb + (((size_t)b * KVH + g) * SS + s) * HDIM;
        o[i]      = f2bf(x1 * cs - x2 * sn);
        o[i + 32] = f2bf(x1 * sn + x2 * cs);
    }
}

// V transpose: qkv f32 [B,S,3072] -> Vt bf16 [B,KV,HD,S]
__global__ void v_transpose(const float* __restrict__ qkv, unsigned short* __restrict__ Vt) {
    int idx = blockIdx.x * blockDim.x + threadIdx.x;
    if (idx >= BB * SS * KVH * HDIM) return;
    int hd = idx & 63;
    int g  = (idx >> 6) % KVH;
    int s  = (idx / (64 * KVH)) % SS;
    int b  = idx / (64 * KVH * SS);
    float v = qkv[((size_t)b * SS + s) * 3072 + 2560 + g * 64 + hd];
    Vt[(((size_t)b * KVH + g) * HDIM + hd) * SS + s] = f2bf(v);
}

// ---------------------------------------------------------------- flash attention (bf16 WMMA)
// Qb [B,H,S,HD], Kb [B,KV,S,HD], Vt [B,KV,HD,S]  ->  Ob bf16 [B,S,H*HD]
__global__ __launch_bounds__(256)
void attn_fwd(const unsigned short* __restrict__ Qb, const unsigned short* __restrict__ Kb,
              const unsigned short* __restrict__ Vt, unsigned short* __restrict__ Ob) {
    __shared__ unsigned int Pl[8][16][18];   // per-wave P tile as (k,k+1) bf16-pair dwords

    const int tid  = threadIdx.x;
    const int lane = tid & 31;
    const int w    = tid >> 5;
    const int l15  = lane & 15;
    const int hi   = lane >> 4;

    const int nqc = SS / 128;                 // 16 query chunks per head
    int blk = blockIdx.x;
    int qc  = blk % nqc;
    int t   = blk / nqc;
    int h   = t % HH;
    int b   = t / HH;
    int g   = h / REPQ;
    int q0  = qc * 128 + w * 16;

    const unsigned short* Qh = Qb + (((size_t)b * HH  + h) * SS) * HDIM;
    const unsigned short* Kh = Kb + (((size_t)b * KVH + g) * SS) * HDIM;
    const unsigned short* Vh = Vt + (((size_t)b * KVH + g) * HDIM) * SS;

    // load Q A-fragments (2 chunks of K=32 over HD=64)
    v8u qf[2];
    {
        int r = q0 + l15;
        const unsigned int* qp = (const unsigned int*)(Qh + (size_t)r * HDIM);
        #pragma unroll
        for (int kc = 0; kc < 2; ++kc) {
            int d0 = kc * 16 + hi * 8;
            #pragma unroll
            for (int v = 0; v < 4; ++v) {
                qf[kc][v]     = qp[d0 + v];
                qf[kc][4 + v] = qp[d0 + 4 + v];
            }
        }
    }

    float mrow[8], lrow[8];
    #pragma unroll
    for (int i = 0; i < 8; ++i) { mrow[i] = -1e30f; lrow[i] = 0.f; }
    v8f accv[4] = {};

    const int ktmax = (q0 + 15) >> 5;
    for (int kt = 0; kt <= ktmax; ++kt) {
        int key0 = kt * 32;

        // prefetch next key tile (K rows and Vt columns) -> global_prefetch_b8
        if (kt < ktmax) {
            __builtin_prefetch(Kh + (size_t)(key0 + 32 + lane) * HDIM, 0, 3);
            __builtin_prefetch(Vh + (size_t)(lane << 1) * SS + key0 + 32, 0, 3);
        }

        // scores: two 16x16 tiles over 32 keys
        v8f sc[2] = {};
        #pragma unroll
        for (int f = 0; f < 2; ++f) {
            int col = key0 + f * 16 + l15;
            const unsigned int* kp = (const unsigned int*)(Kh + (size_t)col * HDIM);
            #pragma unroll
            for (int kc = 0; kc < 2; ++kc) {
                v8u bv;
                int d0 = kc * 16 + hi * 8;
                #pragma unroll
                for (int v = 0; v < 8; ++v) bv[v] = kp[d0 + v];
                sc[f] = wmma_bf16(qf[kc], bv, sc[f]);
            }
        }

        // scale + causal mask + online softmax
        float newm[8], rsum[8], fac[8];
        #pragma unroll
        for (int i = 0; i < 8; ++i) {
            int r  = q0 + hi * 8 + i;
            float s0 = sc[0][i] * 0.125f;
            float s1 = sc[1][i] * 0.125f;
            if (key0 + l15 > r)      s0 = -1e30f;
            if (key0 + 16 + l15 > r) s1 = -1e30f;
            sc[0][i] = s0; sc[1][i] = s1;
            float mx = fmaxf(s0, s1);
            #pragma unroll
            for (int off = 1; off < 16; off <<= 1) mx = fmaxf(mx, __shfl_xor(mx, off));
            newm[i] = fmaxf(mrow[i], mx);
        }
        #pragma unroll
        for (int i = 0; i < 8; ++i) {
            float p0 = __expf(sc[0][i] - newm[i]);
            float p1 = __expf(sc[1][i] - newm[i]);
            int r = hi * 8 + i;
            unsigned short* pw = (unsigned short*)&Pl[w][r][0];
            pw[l15]      = f2bf(p0);
            pw[16 + l15] = f2bf(p1);
            float ps = p0 + p1;
            #pragma unroll
            for (int off = 1; off < 16; off <<= 1) ps += __shfl_xor(ps, off);
            rsum[i] = ps;
            fac[i]  = __expf(mrow[i] - newm[i]);
            lrow[i] = lrow[i] * fac[i] + rsum[i];
            mrow[i] = newm[i];
        }

        // read P back as A-fragment (wave-private LDS; same-wave dep)
        v8u pf;
        {
            int pb = hi * 8;
            #pragma unroll
            for (int v = 0; v < 4; ++v) {
                pf[v]     = Pl[w][l15][pb + v];
                pf[4 + v] = Pl[w][l15][4 + pb + v];
            }
        }

        // P @ V : 4 n-tiles over HD
        #pragma unroll
        for (int nt = 0; nt < 4; ++nt) {
            int col = nt * 16 + l15;
            const unsigned int* vp = (const unsigned int*)(Vh + (size_t)col * SS) + (key0 >> 1);
            v8u bv;
            int d0 = hi * 8;
            #pragma unroll
            for (int v = 0; v < 8; ++v) bv[v] = vp[d0 + v];
            v8f c;
            #pragma unroll
            for (int i = 0; i < 8; ++i) c[i] = accv[nt][i] * fac[i];
            accv[nt] = wmma_bf16(pf, bv, c);
        }
    }

    // epilogue: Ob[b][row][h*64 + hd] bf16
    unsigned short* ob = Ob + ((size_t)b * SS) * (HH * HDIM) + (size_t)h * HDIM
                       + (size_t)(q0 + hi * 8) * (HH * HDIM) + l15;
    #pragma unroll
    for (int i = 0; i < 8; ++i) {
        #pragma unroll
        for (int nt = 0; nt < 4; ++nt) ob[nt * 16] = f2bf(accv[nt][i] / lrow[i]);
        ob += HH * HDIM;
    }
}

// ---------------------------------------------------------------- launch
extern "C" void kernel_launch(void* const* d_in, const int* in_sizes, int n_in,
                              void* d_out, int out_size, void* d_ws, size_t ws_size,
                              hipStream_t stream) {
    const float* x  = (const float*)d_in[0];
    const float* wq = (const float*)d_in[1];
    const float* wk = (const float*)d_in[2];
    const float* wv = (const float*)d_in[3];
    const float* wo = (const float*)d_in[4];
    float* out = (float*)d_out;

    const size_t M = (size_t)BB * SS;           // 4096
    // workspace layout (bytes)
    char* ws = (char*)d_ws;
    unsigned short* xbf   = (unsigned short*)ws;                        ws += M * DD * 2;            // 16 MB
    unsigned short* wqkvb = (unsigned short*)ws;                        ws += (size_t)DD * 3072 * 2; // 12 MB
    unsigned short* wob   = (unsigned short*)ws;                        ws += (size_t)DD * DD * 2;   // 8 MB
    float*          qkv   = (float*)ws;                                 ws += M * 3072 * 4;          // 48 MB
    unsigned short* Qb    = (unsigned short*)ws;                        ws += (size_t)BB * HH  * SS * HDIM * 2; // 16 MB
    unsigned short* Kb    = (unsigned short*)ws;                        ws += (size_t)BB * KVH * SS * HDIM * 2; // 4 MB
    unsigned short* Vt    = (unsigned short*)ws;                        ws += (size_t)BB * KVH * HDIM * SS * 2; // 4 MB
    unsigned short* Ab    = (unsigned short*)ws;                        // attn out bf16, 16 MB

    // 1) casts
    {
        int n = (int)(M * DD);
        cast_f32_bf16<<<(n + 255) / 256, 256, 0, stream>>>(x, xbf, n);
        int nw = DD * 3072;
        pack_wqkv<<<(nw + 255) / 256, 256, 0, stream>>>(wq, wk, wv, wqkvb);
        int no = DD * DD;
        cast_f32_bf16<<<(no + 255) / 256, 256, 0, stream>>>(wo, wob, no);
    }
    // 2) fused QKV projection: [4096,2048] x [2048,3072] -> f32
    {
        dim3 grid((unsigned)(M / BM), 3072 / BN);
        gemm_bf16<<<grid, 256, 0, stream>>>(xbf, wqkvb, qkv, (int)M, 3072, DD);
    }
    // 3) RoPE + V transpose
    {
        int n = BB * SS * (HH + KVH) * 32;
        rope_qk<<<(n + 255) / 256, 256, 0, stream>>>(qkv, Qb, Kb);
        int nv = BB * SS * KVH * HDIM;
        v_transpose<<<(nv + 255) / 256, 256, 0, stream>>>(qkv, Vt);
    }
    // 4) attention
    {
        int blocks = BB * HH * (SS / 128);
        attn_fwd<<<blocks, 256, 0, stream>>>(Qb, Kb, Vt, Ab);
    }
    // 5) output projection: [4096,2048] x [2048,2048] -> f32 d_out
    {
        dim3 grid((unsigned)(M / BM), DD / BN);
        gemm_bf16<<<grid, 256, 0, stream>>>(Ab, wob, out, (int)M, DD, DD);
    }
}
